// VisualModelV_12180527251866
// MI455X (gfx1250) — matrix operand order repaired and checked by reading, not compiled, not run on hardware
//
// ============================================================================
// MI455X (gfx1250) implementation of VisualModelV forward pass.
//
// All GEMMs via v_wmma_f32_16x16x32_f16 (fp32 memory, f16 compute, f32
// accumulate). 64x64x32 LDS-staged block tiles, 4 waves x (32x32) wave tiles
// -> 4 WMMAs per 8 LDS fragment loads. Fused epilogue (bias / strided addend
// / relu / tanh / dual store) makes each recurrent RNN step a single kernel.
// ALIGNED template instantiation gives straight-line staging for the hot
// GEMMs (all have M%64==0, N%64==0, K%32==0); guarded variant only for the
// K=4 box GEMM and the N=10/12 classifier heads. Structured GAT softmax
// (star graph / 54-dense) instead of generic segment ops. Weights (~100MB
// fp32) are L2-resident (192MB).
// ============================================================================
#include <hip/hip_runtime.h>
#include <hip/hip_bf16.h>
#include <math.h>

typedef _Float16 v8h  __attribute__((ext_vector_type(8)));
typedef _Float16 v16h __attribute__((ext_vector_type(16)));
typedef float    v8f  __attribute__((ext_vector_type(8)));

#define TM 64
#define TN 64
#define TK 32
#define KPAD 8   // LDS row pad (halves): keeps 16B alignment, staggers banks

__device__ __forceinline__ v16h cat8(v8h lo, v8h hi) {
  return __builtin_shufflevector(lo, hi, 0,1,2,3,4,5,6,7,8,9,10,11,12,13,14,15);
}

__device__ __forceinline__ void epi_store(float v, int row, int col,
    const float* __restrict__ bias, const float* __restrict__ addend, int ldadd,
    float* __restrict__ C, int ldc, float* __restrict__ C2, int ldc2, int act) {
  if (bias)   v += bias[col];
  if (addend) v += addend[(size_t)row * ldadd + col];
  if (act == 1)      v = fmaxf(v, 0.f);
  else if (act == 2) v = tanhf(v);
  C[(size_t)row * ldc + col] = v;
  if (C2) C2[(size_t)row * ldc2 + col] = v;
}

// C[M,N] = act( A[M,K] * W[N,K]^T + bias[N] + addend[M,N] ).
// 128 threads = 4 waves in a 2x2 grid, each wave owns a 32x32 output tile:
// 2 A fragments x 2 B fragments -> 4 v_wmma per K-step, both fragments reused.
template <bool AL>
__global__ __launch_bounds__(128)
void gemm_nt_wmma(const float* __restrict__ A, int lda,
                  const float* __restrict__ W, int ldw,
                  const float* __restrict__ bias,
                  const float* __restrict__ addend, int ldadd,
                  float* __restrict__ C, int ldc,
                  float* __restrict__ C2, int ldc2,
                  int M, int N, int K, int act) {
  __shared__ _Float16 sA[TM][TK + KPAD];
  __shared__ _Float16 sB[TN][TK + KPAD];

  const int t    = threadIdx.x;
  const int lane = t & 31;
  const int wave = t >> 5;             // 0..3
  const int wm   = (wave >> 1) << 5;   // 0 / 32
  const int wn   = (wave & 1) << 5;    // 0 / 32
  const int l16  = lane & 15;
  const int kh   = lane >> 4;          // K half held by this lane
  const int m0   = blockIdx.y * TM;
  const int n0   = blockIdx.x * TN;

  v8f acc00 = {0.f,0.f,0.f,0.f,0.f,0.f,0.f,0.f};
  v8f acc01 = acc00, acc10 = acc00, acc11 = acc00;

  const int sr = t >> 1;          // staging row 0..63
  const int sc = (t & 1) << 4;    // staging col {0,16}: 16 halves per thread

  for (int k0 = 0; k0 < K; k0 += TK) {
    // ---- stage A tile (f32 -> f16) ----
    if (AL) {
      const float* p = A + (size_t)(m0 + sr) * lda + k0 + sc;
      const float4* q = reinterpret_cast<const float4*>(p);
      float4 x0 = q[0], x1 = q[1], x2 = q[2], x3 = q[3];
      sA[sr][sc+ 0]=(_Float16)x0.x; sA[sr][sc+ 1]=(_Float16)x0.y;
      sA[sr][sc+ 2]=(_Float16)x0.z; sA[sr][sc+ 3]=(_Float16)x0.w;
      sA[sr][sc+ 4]=(_Float16)x1.x; sA[sr][sc+ 5]=(_Float16)x1.y;
      sA[sr][sc+ 6]=(_Float16)x1.z; sA[sr][sc+ 7]=(_Float16)x1.w;
      sA[sr][sc+ 8]=(_Float16)x2.x; sA[sr][sc+ 9]=(_Float16)x2.y;
      sA[sr][sc+10]=(_Float16)x2.z; sA[sr][sc+11]=(_Float16)x2.w;
      sA[sr][sc+12]=(_Float16)x3.x; sA[sr][sc+13]=(_Float16)x3.y;
      sA[sr][sc+14]=(_Float16)x3.z; sA[sr][sc+15]=(_Float16)x3.w;
      if (k0 + TK < K) __builtin_prefetch(p + TK, 0, 1);
    } else {
      int gr = m0 + sr, gc = k0 + sc;
      #pragma unroll
      for (int i = 0; i < 16; ++i) {
        float v = (gr < M && gc + i < K) ? A[(size_t)gr * lda + gc + i] : 0.f;
        sA[sr][sc+i] = (_Float16)v;
      }
    }
    // ---- stage W tile (rows = output features, K-major => acts as B^T) ----
    if (AL) {
      const float* p = W + (size_t)(n0 + sr) * ldw + k0 + sc;
      const float4* q = reinterpret_cast<const float4*>(p);
      float4 x0 = q[0], x1 = q[1], x2 = q[2], x3 = q[3];
      sB[sr][sc+ 0]=(_Float16)x0.x; sB[sr][sc+ 1]=(_Float16)x0.y;
      sB[sr][sc+ 2]=(_Float16)x0.z; sB[sr][sc+ 3]=(_Float16)x0.w;
      sB[sr][sc+ 4]=(_Float16)x1.x; sB[sr][sc+ 5]=(_Float16)x1.y;
      sB[sr][sc+ 6]=(_Float16)x1.z; sB[sr][sc+ 7]=(_Float16)x1.w;
      sB[sr][sc+ 8]=(_Float16)x2.x; sB[sr][sc+ 9]=(_Float16)x2.y;
      sB[sr][sc+10]=(_Float16)x2.z; sB[sr][sc+11]=(_Float16)x2.w;
      sB[sr][sc+12]=(_Float16)x3.x; sB[sr][sc+13]=(_Float16)x3.y;
      sB[sr][sc+14]=(_Float16)x3.z; sB[sr][sc+15]=(_Float16)x3.w;
      if (k0 + TK < K) __builtin_prefetch(p + TK, 0, 1);
    } else {
      int gr = n0 + sr, gc = k0 + sc;
      #pragma unroll
      for (int i = 0; i < 16; ++i) {
        float v = (gr < N && gc + i < K) ? W[(size_t)gr * ldw + gc + i] : 0.f;
        sB[sr][sc+i] = (_Float16)v;
      }
    }
    __syncthreads();

    // B fragments (32x16, lanes 0-15: col=l16 K0..15; lanes 16-31: K16..31)
    v16h bf0 = cat8(*(const v8h*)&sB[wn +      l16][kh * 16],
                    *(const v8h*)&sB[wn +      l16][kh * 16 + 8]);
    v16h bf1 = cat8(*(const v8h*)&sB[wn + 16 + l16][kh * 16],
                    *(const v8h*)&sB[wn + 16 + l16][kh * 16 + 8]);
    // A fragments (16x32, lane half selects K 0..15 vs 16..31)
    v16h af0 = cat8(*(const v8h*)&sA[wm +      l16][kh * 16],
                    *(const v8h*)&sA[wm +      l16][kh * 16 + 8]);
    v16h af1 = cat8(*(const v8h*)&sA[wm + 16 + l16][kh * 16],
                    *(const v8h*)&sA[wm + 16 + l16][kh * 16 + 8]);

    acc00 = __builtin_amdgcn_wmma_f32_16x16x32_f16(false, af0, false, bf0,
                                                   (short)0, acc00, false, false);
    acc01 = __builtin_amdgcn_wmma_f32_16x16x32_f16(false, af0, false, bf1,
                                                   (short)0, acc01, false, false);
    acc10 = __builtin_amdgcn_wmma_f32_16x16x32_f16(false, af1, false, bf0,
                                                   (short)0, acc10, false, false);
    acc11 = __builtin_amdgcn_wmma_f32_16x16x32_f16(false, af1, false, bf1,
                                                   (short)0, acc11, false, false);
    __syncthreads();
  }

  // ---- fused epilogue; C/D layout: vgpr r -> M = r (lanes 0-15) / r+8 ----
  int c0 = n0 + wn + l16;
  int c1 = c0 + 16;
  #pragma unroll
  for (int r = 0; r < 8; ++r) {
    int rr   = (lane < 16) ? r : r + 8;
    int row0 = m0 + wm + rr;
    int row1 = row0 + 16;
    if (AL) {
      epi_store(acc00[r], row0, c0, bias, addend, ldadd, C, ldc, C2, ldc2, act);
      epi_store(acc01[r], row0, c1, bias, addend, ldadd, C, ldc, C2, ldc2, act);
      epi_store(acc10[r], row1, c0, bias, addend, ldadd, C, ldc, C2, ldc2, act);
      epi_store(acc11[r], row1, c1, bias, addend, ldadd, C, ldc, C2, ldc2, act);
    } else {
      if (row0 < M && c0 < N) epi_store(acc00[r], row0, c0, bias, addend, ldadd, C, ldc, C2, ldc2, act);
      if (row0 < M && c1 < N) epi_store(acc01[r], row0, c1, bias, addend, ldadd, C, ldc, C2, ldc2, act);
      if (row1 < M && c0 < N) epi_store(acc10[r], row1, c0, bias, addend, ldadd, C, ldc, C2, ldc2, act);
      if (row1 < M && c1 < N) epi_store(acc11[r], row1, c1, bias, addend, ldadd, C, ldc, C2, ldc2, act);
    }
  }
}

// ---------------------------------------------------------------------------
// Elementwise / data-movement kernels
// ---------------------------------------------------------------------------
__global__ void k_fill(float* p, float v, int n) {
  int i = blockIdx.x * 256 + threadIdx.x;
  if (i < n) p[i] = v;
}

__global__ void k_vec_add(const float* a, const float* b, float* o, int n) {
  int i = blockIdx.x * 256 + threadIdx.x;
  if (i < n) o[i] = a[i] + b[i];
}

__global__ void k_bn_relu(float* x, const float* g, const float* b,
                          const float* m, const float* v, int total, int N) {
  int i = blockIdx.x * 256 + threadIdx.x;
  if (i >= total) return;
  int c = i % N;
  float a = g[c] * rsqrtf(v[c] + 1e-5f);
  x[i] = fmaxf((x[i] - m[c]) * a + b[c], 0.f);
}

// box_input (B,NF,NB,4) -> (B,NB,NF,4)
__global__ void k_transpose_box(const float* in, float* out) {
  int i = blockIdx.x * 256 + threadIdx.x;
  if (i >= 64 * 10 * 6 * 4) return;
  int c  = i & 3;
  int nb = (i >> 2) % 6;
  int f  = ((i >> 2) / 6) % 10;
  int b  = i / 240;
  out[(((b * 6 + nb) * 10 + f) * 4) + c] = in[i];
}

// node[:,0,:, :1280] = feats[:,0];  node[:,1:,:, 1280:] = feats[:,1:]
__global__ void k_scatter_feats(const float* s2, const float* app, float* node) {
  int i = blockIdx.x * 256 + threadIdx.x;
  if (i >= 3840 * 1280) return;
  int row = i / 1280, c = i % 1280;          // rows ordered (b, nb, f)
  int nb = (row / 10) % 6;
  float val = (c < 256) ? s2[(size_t)row * 256 + c]
                        : app[(size_t)row * 1024 + (c - 256)];
  int dc = (nb == 0) ? c : 1280 + c;
  node[(size_t)row * 2560 + dc] = val;
}

// (B,NB,NF,2560) -> (B,NF,NB,2560)
__global__ void k_transpose_node(const float* in, float* out) {
  int i = blockIdx.x * 256 + threadIdx.x;
  if (i >= 3840 * 2560) return;
  int row = i / 2560, c = i % 2560;
  int b = row / 60, nb = (row / 10) % 6, f = row % 10;
  out[(size_t)((b * 10 + f) * 6 + nb) * 2560 + c] = in[(size_t)i];
}

// per-node attention scalars: el = <h, al>, er = <h, ar>  (warp per node)
__global__ void k_el_er(const float* __restrict__ h, const float* al, const float* ar,
                        float* el, float* er, int nodes) {
  int wid  = (blockIdx.x * blockDim.x + threadIdx.x) >> 5;
  int lane = threadIdx.x & 31;
  if (wid >= nodes) return;
  const float* hp = h + (size_t)wid * 512;
  float a = 0.f, b = 0.f;
  for (int c = lane; c < 512; c += 32) { float x = hp[c]; a += x * al[c]; b += x * ar[c]; }
  for (int s = 16; s; s >>= 1) { a += __shfl_xor(a, s, 32); b += __shfl_xor(b, s, 32); }
  if (lane == 0) { el[wid] = a; er[wid] = b; }
}

// Spatial GAT: star graph per (b,f) group. dst nb>0 has one in-edge (node 0);
// dst nb==0 softmaxes over nodes 1..5.
__global__ void k_gat_spatial(const float* __restrict__ h, const float* el,
                              const float* er, const float* bias,
                              float* out, int colOff) {
  int d = blockIdx.x, t = threadIdx.x;
  int nb = d % 6, grp = d - nb;
  float* op = out + (size_t)d * 1024 + colOff;
  if (nb != 0) {
    const float* hp = h + (size_t)grp * 512;
    for (int c = t; c < 512; c += 128) op[c] = fmaxf(hp[c] + bias[c], 0.f);
  } else {
    float e[5], emax = -1e30f;
    #pragma unroll
    for (int i = 0; i < 5; ++i) {
      float x = el[grp + 1 + i] + er[d];
      e[i] = (x > 0.f) ? x : 0.2f * x;
      emax = fmaxf(emax, e[i]);
    }
    float den = 0.f;
    #pragma unroll
    for (int i = 0; i < 5; ++i) { e[i] = expf(e[i] - emax); den += e[i]; }
    float inv = 1.f / den;
    for (int c = t; c < 512; c += 128) {
      float acc = 0.f;
      #pragma unroll
      for (int i = 0; i < 5; ++i) acc += e[i] * h[(size_t)(grp + 1 + i) * 512 + c];
      op[c] = fmaxf(acc * inv + bias[c], 0.f);
    }
  }
}

// Temporal GAT: each dst aggregates all 54 nodes from the other 9 frames.
__global__ void k_gat_temporal(const float* __restrict__ h, const float* el,
                               const float* er, const float* bias,
                               float* out, int colOff) {
  __shared__ float sa[54];
  __shared__ float sred[2];
  int d = blockIdx.x, t = threadIdx.x;
  int f = (d / 6) % 10, b = d / 60;
  if (t < 54) {
    int fi = t / 6, j = t % 6;
    int f2 = fi + (fi >= f ? 1 : 0);
    int src = (b * 10 + f2) * 6 + j;
    float x = el[src] + er[d];
    sa[t] = (x > 0.f) ? x : 0.2f * x;
  }
  __syncthreads();
  if (t == 0) {
    float emax = -1e30f;
    for (int i = 0; i < 54; ++i) emax = fmaxf(emax, sa[i]);
    sred[0] = emax;
  }
  __syncthreads();
  if (t < 54) sa[t] = expf(sa[t] - sred[0]);
  __syncthreads();
  if (t == 0) {
    float den = 0.f;
    for (int i = 0; i < 54; ++i) den += sa[i];
    sred[1] = 1.f / den;
  }
  __syncthreads();
  float inv = sred[1];
  for (int c = t; c < 512; c += 256) {
    float acc = 0.f;
    for (int s = 0; s < 54; ++s) {
      int fi = s / 6, j = s % 6;
      int f2 = fi + (fi >= f ? 1 : 0);
      acc += sa[s] * h[(size_t)((b * 10 + f2) * 6 + j) * 512 + c];
    }
    out[(size_t)d * 1024 + colOff + c] = fmaxf(acc * inv + bias[c], 0.f);
  }
}

__global__ void k_gather_human(const float* st, float* out) {
  int i = blockIdx.x * 256 + threadIdx.x;
  if (i >= 640 * 1024) return;
  int row = i >> 10, c = i & 1023;
  int b = row / 10, f = row % 10;
  out[i] = st[(size_t)((b * 10 + f) * 6) * 1024 + c];
}

__global__ void k_gather_obj(const float* st, float* out) {
  int i = blockIdx.x * 256 + threadIdx.x;
  if (i >= 3200 * 1024) return;
  int row = i >> 10, c = i & 1023;
  int bo = row / 10, f = row % 10;
  int b = bo / 5, o = bo % 5;
  out[i] = st[(size_t)((b * 10 + f) * 6 + (o + 1)) * 1024 + c];
}

__global__ void k_time_sum(const float* z, float* out, int rows, int T, int ncols, int ldz) {
  int i = blockIdx.x * 256 + threadIdx.x;
  if (i >= rows * ncols) return;
  int r = i / ncols, c = i % ncols;
  float s = 0.f;
  for (int t = 0; t < T; ++t) s += z[(size_t)(r * T + t) * ldz + c];
  out[i] = s;
}

// ---------------------------------------------------------------------------
// Host orchestration
// ---------------------------------------------------------------------------
// Input pytree flattened depth-first in dict insertion order (reference file):
enum {
  IN_NUM_OBJS = 0, IN_NODE_FEAT, IN_BOX, IN_BOXCAT,
  IN_SRC_S, IN_DST_S, IN_SRC_T, IN_DST_T,
  P_PRE_W, P_PRE_B,
  P_CW1, P_BN1_G, P_BN1_B, P_BN1_M, P_BN1_V,
  P_CW2, P_BN2_G, P_BN2_B, P_BN2_M, P_BN2_V,
  P_APP_WIHF, P_APP_WHHF, P_APP_BIHF, P_APP_BHHF,
  P_APP_WIHB, P_APP_WHHB, P_APP_BIHB, P_APP_BHHB,
  P_GS_W, P_GS_AL, P_GS_AR, P_GS_B,
  P_GT_W, P_GT_AL, P_GT_AR, P_GT_B,
  P_SUB_WIHF, P_SUB_WHHF, P_SUB_BIHF, P_SUB_BHHF,
  P_SUB_WIHB, P_SUB_WHHB, P_SUB_BIHB, P_SUB_BHHB,
  P_AFF_WIHF, P_AFF_WHHF, P_AFF_BIHF, P_AFF_BHHF,
  P_AFF_WIHB, P_AFF_WHHB, P_AFF_BIHB, P_AFF_BHHB,
  P_CH_W1, P_CH_B1, P_CH_W2, P_CH_B2, P_CH_W3, P_CH_B3,
  P_CO_W1, P_CO_B1, P_CO_W2, P_CO_B2, P_CO_W3, P_CO_B3
};

static inline void gemm(hipStream_t s, const float* A, int lda, const float* W, int ldw,
                        const float* bias, const float* add, int ldadd,
                        float* C, int ldc, float* C2, int ldc2,
                        int M, int N, int K, int act) {
  dim3 g((N + TN - 1) / TN, (M + TM - 1) / TM);
  bool aligned = (M % TM == 0) && (N % TN == 0) && (K % TK == 0);
  if (aligned)
    gemm_nt_wmma<true><<<g, dim3(128), 0, s>>>(A, lda, W, ldw, bias, add, ldadd,
                                               C, ldc, C2, ldc2, M, N, K, act);
  else
    gemm_nt_wmma<false><<<g, dim3(128), 0, s>>>(A, lda, W, ldw, bias, add, ldadd,
                                                C, ldc, C2, ldc2, M, N, K, act);
}

#define EW(kern, n, ...) kern<<<(((n) + 255) / 256), 256, 0, stream>>>(__VA_ARGS__)

extern "C" void kernel_launch(void* const* d_in, const int* in_sizes, int n_in,
                              void* d_out, int out_size, void* d_ws, size_t ws_size,
                              hipStream_t stream) {
  (void)in_sizes; (void)n_in; (void)out_size; (void)ws_size;
  auto F = [&](int i) { return reinterpret_cast<const float*>(d_in[i]); };
  float* out = reinterpret_cast<float*>(d_out);

  float* base = reinterpret_cast<float*>(d_ws);
  size_t off = 0;
  auto alloc = [&](size_t n) { float* p = base + off; off += (n + 63) & ~size_t(63); return p; };

  // ---- workspace ----
  float* biT  = alloc(3840 * 4);
  float* s1   = alloc((size_t)3840 * 128);
  float* s2   = alloc((size_t)3840 * 256);
  float* app  = alloc((size_t)3840 * 1024);
  float* node = alloc((size_t)3840 * 2560);
  float* Gf   = alloc((size_t)3840 * 1280);
  float* Gb   = alloc((size_t)3840 * 1280);
  float* h0a  = alloc((size_t)384 * 1280);
  float* h0b  = alloc((size_t)384 * 1280);
  float* h1a  = alloc((size_t)384 * 1280);
  float* h1b  = alloc((size_t)384 * 1280);
  float* rnnO = alloc((size_t)3840 * 2560);
  float* el_s = alloc(3840); float* er_s = alloc(3840);
  float* el_t = alloc(3840); float* er_t = alloc(3840);
  float* human= alloc((size_t)640 * 1024);
  float* Gsf  = alloc((size_t)640 * 1024);
  float* Gsb  = alloc((size_t)640 * 1024);
  float* hsA  = alloc(64 * 1024); float* hsB = alloc(64 * 1024);
  float* hs2A = alloc(64 * 1024); float* hs2B= alloc(64 * 1024);
  float* hr   = alloc((size_t)640 * 2048);
  float* z1h  = alloc((size_t)640 * 1024);
  float* z2h  = alloc((size_t)640 * 512);
  float* z3h  = alloc((size_t)640 * 10);
  float* haA  = alloc(320 * 1024); float* haB = alloc(320 * 1024);
  float* ha2A = alloc(320 * 1024); float* ha2B= alloc(320 * 1024);
  float* z1o  = alloc((size_t)3200 * 1024);
  float* z2o  = alloc((size_t)3200 * 512);
  float* z3o  = alloc((size_t)3200 * 12);
  float* bAppF= alloc(1280); float* bAppB = alloc(1280);
  float* bSubF= alloc(1024); float* bSubB = alloc(1024);
  float* bAffF= alloc(1024); float* bAffB = alloc(1024);
  // lifetime-safe aliases of the big buffers
  float* nodeT = node;                  // node dead after rnn_app input GEMMs
  float* hs    = Gf;                    // Gf/Gb dead after recurrent loops
  float* ht    = Gf + (size_t)3840 * 512;
  float* st    = Gb;                    // 3840*1024 <= 3840*1280
  float* Gaf   = rnnO;                  // rnnO dead after transpose
  float* Gab   = rnnO + (size_t)3200 * 1024;
  float* orr   = node;                  // nodeT dead after GAT projections

  // ---- stage A: box coordinate MLP + appearance projection ----
  EW(k_transpose_box, 15360, F(IN_BOX), biT);
  gemm(stream, biT, 4, F(P_CW1), 4, nullptr, nullptr, 0, s1, 128, nullptr, 0,
       3840, 128, 4, 0);
  EW(k_bn_relu, 3840 * 128, s1, F(P_BN1_G), F(P_BN1_B), F(P_BN1_M), F(P_BN1_V), 3840 * 128, 128);
  gemm(stream, s1, 128, F(P_CW2), 128, nullptr, nullptr, 0, s2, 256, nullptr, 0,
       3840, 256, 128, 0);
  EW(k_bn_relu, 3840 * 256, s2, F(P_BN2_G), F(P_BN2_B), F(P_BN2_M), F(P_BN2_V), 3840 * 256, 256);
  gemm(stream, F(IN_NODE_FEAT), 2048, F(P_PRE_W), 2048, F(P_PRE_B), nullptr, 0,
       app, 1024, nullptr, 0, 3840, 1024, 2048, 0);
  EW(k_fill, 3840 * 2560, node, 0.f, 3840 * 2560);
  EW(k_scatter_feats, 3840 * 1280, s2, app, node);

  // ---- stage B: rnn_app (batch 384, seq 10, hid 1280/dir) ----
  EW(k_vec_add, 1280, F(P_APP_BIHF), F(P_APP_BHHF), bAppF, 1280);
  EW(k_vec_add, 1280, F(P_APP_BIHB), F(P_APP_BHHB), bAppB, 1280);
  gemm(stream, node, 2560, F(P_APP_WIHF), 2560, bAppF, nullptr, 0,
       Gf, 1280, nullptr, 0, 3840, 1280, 2560, 0);
  gemm(stream, node, 2560, F(P_APP_WIHB), 2560, bAppB, nullptr, 0,
       Gb, 1280, nullptr, 0, 3840, 1280, 2560, 0);
  EW(k_fill, 384 * 1280, h0a, 0.f, 384 * 1280);
  EW(k_fill, 384 * 1280, h1a, 0.f, 384 * 1280);
  {
    float* hin = h0a; float* hout = h0b;
    for (int t = 0; t < 10; ++t) {   // forward
      gemm(stream, hin, 1280, F(P_APP_WHHF), 1280, nullptr, Gf + (size_t)t * 1280, 12800,
           rnnO + (size_t)t * 2560, 25600, hout, 1280, 384, 1280, 1280, 2);
      float* tmp = hin; hin = hout; hout = tmp;
    }
    hin = h1a; hout = h1b;
    for (int s = 0; s < 10; ++s) {   // backward
      int t = 9 - s;
      gemm(stream, hin, 1280, F(P_APP_WHHB), 1280, nullptr, Gb + (size_t)t * 1280, 12800,
           rnnO + (size_t)t * 2560 + 1280, 25600, hout, 1280, 384, 1280, 1280, 2);
      float* tmp = hin; hin = hout; hout = tmp;
    }
  }
  EW(k_transpose_node, 3840 * 2560, rnnO, nodeT);   // (b,nb,f) -> (b,f,nb)

  // ---- stage C: two GATs ----
  gemm(stream, nodeT, 2560, F(P_GS_W), 2560, nullptr, nullptr, 0,
       hs, 512, nullptr, 0, 3840, 512, 2560, 0);
  gemm(stream, nodeT, 2560, F(P_GT_W), 2560, nullptr, nullptr, 0,
       ht, 512, nullptr, 0, 3840, 512, 2560, 0);
  k_el_er<<<480, 256, 0, stream>>>(hs, F(P_GS_AL), F(P_GS_AR), el_s, er_s, 3840);
  k_el_er<<<480, 256, 0, stream>>>(ht, F(P_GT_AL), F(P_GT_AR), el_t, er_t, 3840);
  k_gat_spatial<<<3840, 128, 0, stream>>>(hs, el_s, er_s, F(P_GS_B), st, 0);
  k_gat_temporal<<<3840, 256, 0, stream>>>(ht, el_t, er_t, F(P_GT_B), st, 512);

  // ---- stage D: split human/object streams + their BiRNNs ----
  EW(k_gather_human, 640 * 1024, st, human);
  float* obj = app;                    // app dead after scatter_feats
  EW(k_gather_obj, 3200 * 1024, st, obj);

  EW(k_vec_add, 1024, F(P_SUB_BIHF), F(P_SUB_BHHF), bSubF, 1024);
  EW(k_vec_add, 1024, F(P_SUB_BIHB), F(P_SUB_BHHB), bSubB, 1024);
  EW(k_vec_add, 1024, F(P_AFF_BIHF), F(P_AFF_BHHF), bAffF, 1024);
  EW(k_vec_add, 1024, F(P_AFF_BIHB), F(P_AFF_BHHB), bAffB, 1024);

  gemm(stream, human, 1024, F(P_SUB_WIHF), 1024, bSubF, nullptr, 0,
       Gsf, 1024, nullptr, 0, 640, 1024, 1024, 0);
  gemm(stream, human, 1024, F(P_SUB_WIHB), 1024, bSubB, nullptr, 0,
       Gsb, 1024, nullptr, 0, 640, 1024, 1024, 0);
  EW(k_fill, 64 * 1024, hsA, 0.f, 64 * 1024);
  EW(k_fill, 64 * 1024, hs2A, 0.f, 64 * 1024);
  {
    float* hin = hsA; float* hout = hsB;
    for (int t = 0; t < 10; ++t) {
      gemm(stream, hin, 1024, F(P_SUB_WHHF), 1024, nullptr, Gsf + (size_t)t * 1024, 10240,
           hr + (size_t)t * 2048, 20480, hout, 1024, 64, 1024, 1024, 2);
      float* tmp = hin; hin = hout; hout = tmp;
    }
    hin = hs2A; hout = hs2B;
    for (int s = 0; s < 10; ++s) {
      int t = 9 - s;
      gemm(stream, hin, 1024, F(P_SUB_WHHB), 1024, nullptr, Gsb + (size_t)t * 1024, 10240,
           hr + (size_t)t * 2048 + 1024, 20480, hout, 1024, 64, 1024, 1024, 2);
      float* tmp = hin; hin = hout; hout = tmp;
    }
  }

  gemm(stream, obj, 1024, F(P_AFF_WIHF), 1024, bAffF, nullptr, 0,
       Gaf, 1024, nullptr, 0, 3200, 1024, 1024, 0);
  gemm(stream, obj, 1024, F(P_AFF_WIHB), 1024, bAffB, nullptr, 0,
       Gab, 1024, nullptr, 0, 3200, 1024, 1024, 0);
  EW(k_fill, 320 * 1024, haA, 0.f, 320 * 1024);
  EW(k_fill, 320 * 1024, ha2A, 0.f, 320 * 1024);
  {
    float* hin = haA; float* hout = haB;
    for (int t = 0; t < 10; ++t) {
      gemm(stream, hin, 1024, F(P_AFF_WHHF), 1024, nullptr, Gaf + (size_t)t * 1024, 10240,
           orr + (size_t)t * 2048, 20480, hout, 1024, 320, 1024, 1024, 2);
      float* tmp = hin; hin = hout; hout = tmp;
    }
    hin = ha2A; hout = ha2B;
    for (int s = 0; s < 10; ++s) {
      int t = 9 - s;
      gemm(stream, hin, 1024, F(P_AFF_WHHB), 1024, nullptr, Gab + (size_t)t * 1024, 10240,
           orr + (size_t)t * 2048 + 1024, 20480, hout, 1024, 320, 1024, 1024, 2);
      float* tmp = hin; hin = hout; hout = tmp;
    }
  }

  // ---- stage E: classifier MLPs + time-sum ----
  gemm(stream, hr, 2048, F(P_CH_W1), 2048, F(P_CH_B1), nullptr, 0,
       z1h, 1024, nullptr, 0, 640, 1024, 2048, 1);
  gemm(stream, z1h, 1024, F(P_CH_W2), 1024, F(P_CH_B2), nullptr, 0,
       z2h, 512, nullptr, 0, 640, 512, 1024, 1);
  gemm(stream, z2h, 512, F(P_CH_W3), 512, F(P_CH_B3), nullptr, 0,
       z3h, 10, nullptr, 0, 640, 10, 512, 0);
  EW(k_time_sum, 64 * 10, z3h, out, 64, 10, 10, 10);

  gemm(stream, orr, 2048, F(P_CO_W1), 2048, F(P_CO_B1), nullptr, 0,
       z1o, 1024, nullptr, 0, 3200, 1024, 2048, 1);
  gemm(stream, z1o, 1024, F(P_CO_W2), 1024, F(P_CO_B2), nullptr, 0,
       z2o, 512, nullptr, 0, 3200, 512, 1024, 1);
  gemm(stream, z2o, 512, F(P_CO_W3), 512, F(P_CO_B3), nullptr, 0,
       z3o, 12, nullptr, 0, 3200, 12, 512, 0);
  EW(k_time_sum, 320 * 12, z3o, out + 640, 320, 10, 12, 12);
}
